// HybridClassicalQuantumAutoencoder_31679678775407
// MI455X (gfx1250) — compile-verified, matrix operand order checked by reading, not run on hardware
//
#include <hip/hip_runtime.h>
#include <math.h>

// ---------------------------------------------------------------------------
// Hybrid classical-quantum autoencoder forward pass for gfx1250 (MI455X).
// GEMMs use V_WMMA_F32_16X16X32_BF16 (bf16 in, f32 accumulate); all GEMM
// shapes are compile-time constants so inner-loop addressing is pure
// integer-offset increment (no 64-bit multiplies), and addresses are formed
// as GEPs on the kernarg pointers so loads stay GLOBAL (not FLAT).
// Quantum 8-qubit statevector sim runs one workgroup per batch element with
// the 256-amplitude complex state resident in LDS.
// ---------------------------------------------------------------------------

typedef __attribute__((ext_vector_type(16))) __bf16 v16bf;
typedef __attribute__((ext_vector_type(8)))  float  v8f;

#define B_SZ    4096
#define IN_DIM  2560
#define ENC_DIM 256

__device__ __forceinline__ unsigned short f2bf(float f) {
  unsigned int u = __float_as_uint(f);
  u += 0x7FFFu + ((u >> 16) & 1u);           // round-to-nearest-even
  return (unsigned short)(u >> 16);
}

__device__ __forceinline__ void store_out(float* p, size_t i, float v) { p[i] = v; }
__device__ __forceinline__ void store_out(unsigned short* p, size_t i, float v) { p[i] = f2bf(v); }

// ---------------- conversion kernels ----------------
__global__ void k_f32_to_bf16(const float* __restrict__ s,
                              unsigned short* __restrict__ d, int n) {
  int i = blockIdx.x * blockDim.x + threadIdx.x;
  if (i < n) d[i] = f2bf(s[i]);
}

// Convert with row zero-padding (for dw1: 24x128 -> 32x128 so K becomes 32).
__global__ void k_f32_to_bf16_padrows(const float* __restrict__ s,
                                      unsigned short* __restrict__ d,
                                      int rows_src, int rows_dst, int cols) {
  int i = blockIdx.x * blockDim.x + threadIdx.x;
  int n = rows_dst * cols;
  if (i < n) {
    int r = i / cols;
    d[i] = (r < rows_src) ? f2bf(s[i]) : (unsigned short)0;
  }
}

// ---------------- WMMA GEMM ----------------
// C[M,N] = act(A[M,K] @ B[K,N] + bias), A/B bf16 row-major, acc f32.
// Block tile 128x128, BK=32, 256 threads (8 waves), wave tile 32x64.
#define BM 128
#define BN 128
#define BK 32
#define LSTR 40          // LDS row stride in bf16 elements (80B, 16B aligned)

enum { ACT_LEAKY = 0, ACT_TANH = 1 };

union Frag16 { unsigned int u[8]; v16bf v; };
union Chunk  { uint4 q; unsigned short h[8]; };

template <int ACT, typename OutT, int N, int K>
__global__ __launch_bounds__(256) void k_gemm_bf16(
    const unsigned short* __restrict__ A,
    const unsigned short* __restrict__ B,
    const float* __restrict__ bias,
    OutT* __restrict__ C) {
  __shared__ unsigned short As[BM * LSTR];   // As[m][k]
  __shared__ unsigned short Bs[BN * LSTR];   // Bs[n][k]  (transposed tile)

  const int tid  = threadIdx.x;
  const int lane = tid & 31;
  const int wave = tid >> 5;
  const int wr   = wave >> 1;      // 0..3 : 32-row group
  const int wc   = wave & 1;       // 0..1 : 64-col group
  const int nl   = lane & 15;
  const int half = lane >> 4;

  const int row0 = blockIdx.y * BM;
  const int col0 = blockIdx.x * BN;

  // global->LDS chunk coordinates: 512 chunks of 8 halfs per tile, 2/thread
  int arow[2], acol[2], bkr[2], bnc[2];
#pragma unroll
  for (int j = 0; j < 2; ++j) {
    int c = tid + j * 256;
    arow[j] = c >> 2;  acol[j] = (c & 3) * 8;    // A: 128 rows x 4 chunks
    bkr[j]  = c >> 4;  bnc[j]  = (c & 15) * 8;   // B: 32 rows x 16 chunks
  }

  // Loop-carried INTEGER offsets (constant-stride increments); addresses are
  // always fresh GEPs on the kernarg pointers so they stay in addrspace(1).
  size_t aOff[2], bOff[2];
#pragma unroll
  for (int j = 0; j < 2; ++j) {
    aOff[j] = (size_t)(row0 + arow[j]) * K + acol[j];
    bOff[j] = (size_t)bkr[j] * N + col0 + bnc[j];
  }

  // per-lane K offsets for 16-bit WMMA fragments (ISA 7.12.2 layout):
  // VGPR v holds packed (k, k+1), k = 2*(v&3) + 16*(v>=4) + 8*laneHalf
  int koff[8];
#pragma unroll
  for (int v = 0; v < 8; ++v) koff[v] = (v & 3) * 2 + (v >> 2) * 16 + half * 8;

  v8f acc[2][4];
#pragma unroll
  for (int mi = 0; mi < 2; ++mi)
#pragma unroll
    for (int ni = 0; ni < 4; ++ni)
#pragma unroll
      for (int r = 0; r < 8; ++r) acc[mi][ni][r] = 0.0f;

  constexpr int KT = K / BK;
  constexpr size_t ASTEP = BK;
  constexpr size_t BSTEP = (size_t)BK * N;

  Chunk aS[2], bS[2];
  // preload tile 0 into registers, then LDS
#pragma unroll
  for (int j = 0; j < 2; ++j) {
    aS[j].q = *(const uint4*)(A + aOff[j]);
    bS[j].q = *(const uint4*)(B + bOff[j]);
    aOff[j] += ASTEP;
    bOff[j] += BSTEP;
  }
#pragma unroll
  for (int j = 0; j < 2; ++j) {
    *(uint4*)&As[arow[j] * LSTR + acol[j]] = aS[j].q;
#pragma unroll
    for (int e = 0; e < 8; ++e) Bs[(bnc[j] + e) * LSTR + bkr[j]] = bS[j].h[e];
  }
  __syncthreads();

  for (int kt = 0; kt < KT; ++kt) {
    // stage next tile in registers while WMMAs run on current LDS tile
    if (kt + 1 < KT) {
#pragma unroll
      for (int j = 0; j < 2; ++j) {
        aS[j].q = *(const uint4*)(A + aOff[j]);
        bS[j].q = *(const uint4*)(B + bOff[j]);
      }
      if (kt + 2 < KT) {
        __builtin_prefetch(A + aOff[0] + ASTEP, 0, 3);   // near-cache prefetch
        __builtin_prefetch(B + bOff[0] + BSTEP, 0, 3);
      }
#pragma unroll
      for (int j = 0; j < 2; ++j) {
        aOff[j] += ASTEP;
        bOff[j] += BSTEP;
      }
    }

    Frag16 af[2];
#pragma unroll
    for (int mi = 0; mi < 2; ++mi) {
      const int rb = (wr * 32 + mi * 16 + nl) * LSTR;
#pragma unroll
      for (int v = 0; v < 8; ++v)
        af[mi].u[v] = *(const unsigned int*)&As[rb + koff[v]];
    }
#pragma unroll
    for (int ni = 0; ni < 4; ++ni) {
      Frag16 bfr;
      const int cb = (wc * 64 + ni * 16 + nl) * LSTR;
#pragma unroll
      for (int v = 0; v < 8; ++v)
        bfr.u[v] = *(const unsigned int*)&Bs[cb + koff[v]];
#pragma unroll
      for (int mi = 0; mi < 2; ++mi)
        acc[mi][ni] = __builtin_amdgcn_wmma_f32_16x16x32_bf16(
            false, af[mi].v, false, bfr.v, (short)0, acc[mi][ni], false, false);
    }

    __syncthreads();
    if (kt + 1 < KT) {
#pragma unroll
      for (int j = 0; j < 2; ++j) {
        *(uint4*)&As[arow[j] * LSTR + acol[j]] = aS[j].q;
#pragma unroll
        for (int e = 0; e < 8; ++e) Bs[(bnc[j] + e) * LSTR + bkr[j]] = bS[j].h[e];
      }
    }
    __syncthreads();
  }

  // epilogue: f32 C/D layout -> VGPR r holds (M = r + 8*half, N = lane&15)
#pragma unroll
  for (int mi = 0; mi < 2; ++mi) {
#pragma unroll
    for (int ni = 0; ni < 4; ++ni) {
      const int gm0 = row0 + wr * 32 + mi * 16 + half * 8;
      const int gn  = col0 + wc * 64 + ni * 16 + nl;
      const float bv = bias[gn];
#pragma unroll
      for (int r = 0; r < 8; ++r) {
        float x = acc[mi][ni][r] + bv;
        if (ACT == ACT_LEAKY) x = (x >= 0.0f) ? x : 0.2f * x;
        else                  x = tanhf(x);
        store_out(C, (size_t)(gm0 + r) * N + gn, x);
      }
    }
  }
}

// ---------------- 8-qubit statevector kernel ----------------
// One block per batch element; thread i owns amplitude i (qubit q <-> bit 7-q).
__global__ __launch_bounds__(256) void k_quantum(
    const float* __restrict__ enc, const float* __restrict__ par,
    unsigned short* __restrict__ meas /* B_SZ x 32 bf16, cols 24..31 zero */) {
  __shared__ float pr[256];
  __shared__ float pim[256];
  __shared__ float red[256];
  const int tid = threadIdx.x;
  const int b = blockIdx.x;

  // normalize encoder row -> real initial state
  const float v = enc[(size_t)b * 256 + tid];
  red[tid] = v * v;
  __syncthreads();
#pragma unroll
  for (int s = 128; s > 0; s >>= 1) { if (tid < s) red[tid] += red[tid + s]; __syncthreads(); }
  const float inv = 1.0f / fmaxf(sqrtf(red[0]), 1e-12f);
  __syncthreads();
  pr[tid] = v * inv;
  pim[tid] = 0.0f;
  __syncthreads();

  for (int l = 0; l < 6; ++l) {
    // Rot(phi, theta, omega) = RZ(omega) RY(theta) RZ(phi) on each qubit
    for (int q = 0; q < 8; ++q) {
      const float phi = par[(l * 8 + q) * 3 + 0];
      const float th  = par[(l * 8 + q) * 3 + 1];
      const float om  = par[(l * 8 + q) * 3 + 2];
      float st, ct;  __sincosf(0.5f * th, &st, &ct);
      float sp, cp;  __sincosf(0.5f * (phi + om), &sp, &cp);
      float sm, cm;  __sincosf(0.5f * (phi - om), &sm, &cm);
      const float m00r =  cp * ct, m00i = -sp * ct;
      const float m01r = -cm * st, m01i = -sm * st;
      const float m10r =  cm * st, m10i = -sm * st;
      const float m11r =  cp * ct, m11i =  sp * ct;
      const int p = 7 - q;
      const int msk = 1 << p;
      const int i0 = tid & ~msk, i1 = tid | msk;
      const int bit = (tid >> p) & 1;
      const float a0r = pr[i0], a0i = pim[i0], a1r = pr[i1], a1i = pim[i1];
      __syncthreads();
      float nr_, ni_;
      if (!bit) {
        nr_ = m00r * a0r - m00i * a0i + m01r * a1r - m01i * a1i;
        ni_ = m00r * a0i + m00i * a0r + m01r * a1i + m01i * a1r;
      } else {
        nr_ = m10r * a0r - m10i * a0i + m11r * a1r - m11i * a1i;
        ni_ = m10r * a0i + m10i * a0r + m11r * a1i + m11i * a1r;
      }
      pr[tid] = nr_; pim[tid] = ni_;
      __syncthreads();
    }
    // CNOT ring with range r = l % 7 + 1
    const int rr = l % 7 + 1;
    for (int q = 0; q < 8; ++q) {
      const int pc = 7 - q;
      const int pt = 7 - ((q + rr) & 7);
      const int src = ((tid >> pc) & 1) ? (tid ^ (1 << pt)) : tid;
      const float sr = pr[src], si = pim[src];
      __syncthreads();
      pr[tid] = sr; pim[tid] = si;
      __syncthreads();
    }
  }

  // measurements: <Z_q>, <X_q>, <Y_q>
  const float ar = pr[tid], ai = pim[tid];
  const float p2 = ar * ar + ai * ai;
  for (int q = 0; q < 8; ++q) {
    const int p = 7 - q;
    const int j = tid ^ (1 << p);
    const int bit = (tid >> p) & 1;
    const float br = pr[j], bi = pim[j];
    const float zc = bit ? -p2 : p2;
    const float xc = ar * br + ai * bi;            // Re(conj(a)*b)
    const float ci = ar * bi - ai * br;            // Im(conj(a)*b)
    const float yc = bit ? -ci : ci;               // Re(conj(a)*b * (+/-i))

    red[tid] = zc; __syncthreads();
    for (int s = 128; s > 0; s >>= 1) { if (tid < s) red[tid] += red[tid + s]; __syncthreads(); }
    const float Z = red[0]; __syncthreads();
    red[tid] = xc; __syncthreads();
    for (int s = 128; s > 0; s >>= 1) { if (tid < s) red[tid] += red[tid + s]; __syncthreads(); }
    const float X = red[0]; __syncthreads();
    red[tid] = yc; __syncthreads();
    for (int s = 128; s > 0; s >>= 1) { if (tid < s) red[tid] += red[tid + s]; __syncthreads(); }
    const float Y = red[0]; __syncthreads();

    if (tid == 0) {
      meas[(size_t)b * 32 + q]      = f2bf(Z);
      meas[(size_t)b * 32 + 8 + q]  = f2bf(X);
      meas[(size_t)b * 32 + 16 + q] = f2bf(Y);
    }
  }
  if (tid < 8) meas[(size_t)b * 32 + 24 + tid] = 0;  // K-pad for decoder GEMM
}

// ---------------------------------------------------------------------------
extern "C" void kernel_launch(void* const* d_in, const int* in_sizes, int n_in,
                              void* d_out, int out_size, void* d_ws, size_t ws_size,
                              hipStream_t stream) {
  (void)in_sizes; (void)n_in; (void)out_size; (void)ws_size;

  const float* x    = (const float*)d_in[0];
  const float* ew1  = (const float*)d_in[1];
  const float* eb1  = (const float*)d_in[2];
  const float* ew2  = (const float*)d_in[3];
  const float* eb2  = (const float*)d_in[4];
  const float* ew3  = (const float*)d_in[5];
  const float* eb3  = (const float*)d_in[6];
  const float* dpar = (const float*)d_in[7];
  const float* dw1  = (const float*)d_in[8];
  const float* db1  = (const float*)d_in[9];
  const float* dw2  = (const float*)d_in[10];
  const float* db2  = (const float*)d_in[11];
  const float* dw3  = (const float*)d_in[12];
  const float* db3  = (const float*)d_in[13];
  const float* dw4  = (const float*)d_in[14];
  const float* db4  = (const float*)d_in[15];
  const float* dw5  = (const float*)d_in[16];
  const float* db5  = (const float*)d_in[17];
  float* out = (float*)d_out;

  // workspace carve-out (~67 MB total), 256B aligned slices
  char* ws = (char*)d_ws;
  size_t off = 0;
  auto carve = [&](size_t bytes) -> void* {
    void* p = ws + off;
    off += (bytes + 255) & ~(size_t)255;
    return p;
  };
  unsigned short* xb   = (unsigned short*)carve((size_t)B_SZ * IN_DIM * 2);
  unsigned short* w1b  = (unsigned short*)carve((size_t)IN_DIM * 1024 * 2);
  unsigned short* w2b  = (unsigned short*)carve((size_t)1024 * 512 * 2);
  unsigned short* w3b  = (unsigned short*)carve((size_t)512 * 256 * 2);
  unsigned short* v1b  = (unsigned short*)carve((size_t)32 * 128 * 2);      // dw1 padded
  unsigned short* v2b  = (unsigned short*)carve((size_t)128 * 256 * 2);
  unsigned short* v3b  = (unsigned short*)carve((size_t)256 * 512 * 2);
  unsigned short* v4b  = (unsigned short*)carve((size_t)512 * 1024 * 2);
  unsigned short* v5b  = (unsigned short*)carve((size_t)1024 * IN_DIM * 2);
  unsigned short* h1b  = (unsigned short*)carve((size_t)B_SZ * 1024 * 2);
  unsigned short* h2b  = (unsigned short*)carve((size_t)B_SZ * 512 * 2);
  float*          encf = (float*)carve((size_t)B_SZ * 256 * 4);
  unsigned short* msb  = (unsigned short*)carve((size_t)B_SZ * 32 * 2);
  unsigned short* d1b  = (unsigned short*)carve((size_t)B_SZ * 128 * 2);
  unsigned short* d2b  = (unsigned short*)carve((size_t)B_SZ * 256 * 2);
  unsigned short* d3b  = (unsigned short*)carve((size_t)B_SZ * 512 * 2);
  unsigned short* d4b  = (unsigned short*)carve((size_t)B_SZ * 1024 * 2);

  auto conv = [&](const float* s, unsigned short* d, int n) {
    k_f32_to_bf16<<<(n + 255) / 256, 256, 0, stream>>>(s, d, n);
  };
  conv(x,   xb,  B_SZ * IN_DIM);
  conv(ew1, w1b, IN_DIM * 1024);
  conv(ew2, w2b, 1024 * 512);
  conv(ew3, w3b, 512 * 256);
  k_f32_to_bf16_padrows<<<(32 * 128 + 255) / 256, 256, 0, stream>>>(dw1, v1b, 24, 32, 128);
  conv(dw2, v2b, 128 * 256);
  conv(dw3, v3b, 256 * 512);
  conv(dw4, v4b, 512 * 1024);
  conv(dw5, v5b, 1024 * IN_DIM);

  // encoder
  k_gemm_bf16<ACT_LEAKY, unsigned short, 1024, IN_DIM>
      <<<dim3(1024 / BN, B_SZ / BM), 256, 0, stream>>>(xb,  w1b, eb1, h1b);
  k_gemm_bf16<ACT_LEAKY, unsigned short, 512, 1024>
      <<<dim3(512 / BN, B_SZ / BM), 256, 0, stream>>>(h1b, w2b, eb2, h2b);
  k_gemm_bf16<ACT_TANH, float, 256, 512>
      <<<dim3(256 / BN, B_SZ / BM), 256, 0, stream>>>(h2b, w3b, eb3, encf);

  // quantum statevector circuit + measurements
  k_quantum<<<B_SZ, 256, 0, stream>>>(encf, dpar, msb);

  // decoder
  k_gemm_bf16<ACT_LEAKY, unsigned short, 128, 32>
      <<<dim3(128 / BN, B_SZ / BM), 256, 0, stream>>>(msb, v1b, db1, d1b);
  k_gemm_bf16<ACT_LEAKY, unsigned short, 256, 128>
      <<<dim3(256 / BN, B_SZ / BM), 256, 0, stream>>>(d1b, v2b, db2, d2b);
  k_gemm_bf16<ACT_LEAKY, unsigned short, 512, 256>
      <<<dim3(512 / BN, B_SZ / BM), 256, 0, stream>>>(d2b, v3b, db3, d3b);
  k_gemm_bf16<ACT_LEAKY, unsigned short, 1024, 512>
      <<<dim3(1024 / BN, B_SZ / BM), 256, 0, stream>>>(d3b, v4b, db4, d4b);
  k_gemm_bf16<ACT_TANH, float, IN_DIM, 1024>
      <<<dim3(IN_DIM / BN, B_SZ / BM), 256, 0, stream>>>(d4b, v5b, db5, out);
}
